// Qwen3Attention_60851096649884
// MI455X (gfx1250) — matrix operand order repaired and checked by reading, not compiled, not run on hardware
//
#include <hip/hip_runtime.h>
#include <cstdint>

// ---- problem constants (from setup_inputs) ----
#define BATCH    2
#define QLEN     1024
#define NH       32
#define KVH      32
#define HDIM     128
#define BS       16
#define MAXBLK   (QLEN / BS)      // 64 blocks per sequence
#define TILE_Q   128              // q rows per workgroup (8 waves x 16)
#define TILE_KV  32               // kv positions per iteration (2 cache blocks)
#define NWAVES   8
#define VT_STRIDE 40              // padded kv-stride of transposed V tile (bank-conflict relief)

typedef __attribute__((ext_vector_type(8)))  float          v8f;
typedef __attribute__((ext_vector_type(4)))  float          v4f;
typedef __attribute__((ext_vector_type(16))) unsigned short v16u;
typedef __attribute__((ext_vector_type(8)))  unsigned short v8u;
typedef __attribute__((ext_vector_type(2)))  unsigned int   v2ui;
typedef __attribute__((ext_vector_type(16))) __bf16         v16bf;

static __device__ __forceinline__ unsigned short f2bf(float f) {
    union { float f; unsigned int u; } c; c.f = f;
    unsigned int r = c.u + 0x7FFFu + ((c.u >> 16) & 1u);   // round-nearest-even
    return (unsigned short)(r >> 16);
}

static __device__ __forceinline__ v16bf as_bf(v16u v) {
    union { v16u u; v16bf b; } c; c.u = v; return c.b;
}

static __device__ __forceinline__ v2ui pack4bf(v4f f) {
    v2ui r;
    r[0] = (unsigned)f2bf(f[0]) | ((unsigned)f2bf(f[1]) << 16);
    r[1] = (unsigned)f2bf(f[2]) | ((unsigned)f2bf(f[3]) << 16);
    return r;
}

// ds_swizzle_b32 with immediate XOR pattern (group-of-32: {xor[14:10], or[9:5], and[4:0]})
template <int PATTERN>
static __device__ __forceinline__ float swzf(float v) {
    return __uint_as_float(
        (unsigned)__builtin_amdgcn_ds_swizzle((int)__float_as_uint(v), PATTERN));
}
static __device__ __forceinline__ float rowmax16(float v) {
    v = fmaxf(v, swzf<0x041F>(v));   // xor 1
    v = fmaxf(v, swzf<0x081F>(v));   // xor 2
    v = fmaxf(v, swzf<0x101F>(v));   // xor 4
    v = fmaxf(v, swzf<0x201F>(v));   // xor 8
    return v;
}
static __device__ __forceinline__ float rowsum16(float v) {
    v += swzf<0x041F>(v);
    v += swzf<0x081F>(v);
    v += swzf<0x101F>(v);
    v += swzf<0x201F>(v);
    return v;
}

__global__ __launch_bounds__(256)
void qwen3_pa_fwd(const float* __restrict__ q,
                  const float* __restrict__ k_cache,
                  const float* __restrict__ v_cache,
                  const int*   __restrict__ block_tables,
                  float*       __restrict__ out) {
    // double-buffered K tile: [buf][kvLocal][d]  row-major bf16
    __shared__ __attribute__((aligned(32))) unsigned short sK[2][TILE_KV * HDIM];      // 16 KB
    // double-buffered V tile transposed: [buf][d][kvLocal] bf16, padded stride
    __shared__ __attribute__((aligned(32))) unsigned short sVt[2][HDIM * VT_STRIDE];   // 20 KB
    // per-wave P scratch: [m][k] 16x32 bf16
    __shared__ __attribute__((aligned(32))) unsigned short sP[NWAVES][16 * TILE_KV];   // 8 KB

    const int tid  = threadIdx.x;
    const int wave = __builtin_amdgcn_readfirstlane(tid >> 5);
    const int lane = tid & 31;
    const int lh   = lane >> 4;    // lane half (0/1)
    const int ln   = lane & 15;    // lane-in-half

    const int qTile = blockIdx.x & 7;           // QLEN/TILE_Q = 8
    const int bh    = blockIdx.x >> 3;
    const int h     = bh % NH;
    const int b     = bh / NH;
    const int qBase = qTile * TILE_Q;

    const float scale = 0.08838834764831845f;   // 1/sqrt(128)

    // staging geometry: 32 threads cover one 128-float row (float4 each); 8 row-groups
    const int d0   = (tid & 31) * 4;            // this thread's 4 d-columns
    const int rowG = tid >> 5;                  // kv rows rowG + 8*i, i=0..3

    // ---- load Q fragments (A layout, 16x32 bf16 per D-chunk), pre-scaled ----
    const int   tokQ = b * QLEN + qBase + wave * 16 + ln;   // A layout: m = ln
    const float* qrow = q + ((size_t)tokQ * NH + h) * HDIM;
    v16u aQ[4];
#pragma unroll
    for (int c = 0; c < 4; ++c) {
#pragma unroll
        for (int e = 0; e < 16; ++e) {
            const int k = lh * 16 + e;   // A-matrix K index
            aQ[c][e] = f2bf(qrow[c * 32 + k] * scale);
        }
    }

    // ---- accumulators & softmax state (C layout: row m = lh*8 + e) ----
    v8f   oAcc[8];
#pragma unroll
    for (int j = 0; j < 8; ++j) oAcc[j] = {};
    float mRow[8], lRow[8];
#pragma unroll
    for (int e = 0; e < 8; ++e) { mRow[e] = -INFINITY; lRow[e] = 0.0f; }

    const int nTiles  = (qBase + TILE_Q) / TILE_KV;
    const int qRowMax = qBase + wave * 16 + 15;
    const int qRowMin = qBase + wave * 16;

    // ---- prologue: stage tile 0 into buffer 0 ----
    {
#pragma unroll
        for (int i = 0; i < 4; ++i) {
            const int kv  = rowG + 8 * i;
            const int bt  = block_tables[b * MAXBLK + (kv >> 4)];
            const size_t src = (((size_t)bt * KVH + h) * BS + (kv & (BS - 1))) * HDIM + d0;
            const v4f kd = *(const v4f*)(k_cache + src);
            const v4f vd = *(const v4f*)(v_cache + src);
            *(v2ui*)&sK[0][kv * HDIM + d0] = pack4bf(kd);
#pragma unroll
            for (int dd = 0; dd < 4; ++dd)
                sVt[0][(d0 + dd) * VT_STRIDE + kv] = f2bf(vd[dd]);
        }
    }
    __syncthreads();

    for (int it = 0; it < nTiles; ++it) {
        const int kvBase = it * TILE_KV;
        const int p      = it & 1;
        const bool haveNext = (it + 1) < nTiles;

        // ---- issue next tile's global loads (fly under the WMMAs below) ----
        v4f kN[4], vN[4];
        if (haveNext) {
            const int kvBaseN = kvBase + TILE_KV;
#pragma unroll
            for (int i = 0; i < 4; ++i) {
                const int pos = kvBaseN + rowG + 8 * i;
                const int bt  = block_tables[b * MAXBLK + (pos >> 4)];
                const size_t src = (((size_t)bt * KVH + h) * BS + (pos & (BS - 1))) * HDIM + d0;
                kN[i] = *(const v4f*)(k_cache + src);
                vN[i] = *(const v4f*)(v_cache + src);
            }
            // prefetch tile it+2 into L2
            if (it + 2 < nTiles) {
                const int pn  = kvBase + 2 * TILE_KV + rowG;
                const int btn = block_tables[b * MAXBLK + (pn >> 4)];
                const size_t srcn = (((size_t)btn * KVH + h) * BS + (pn & 15)) * HDIM + d0;
                __builtin_prefetch(&k_cache[srcn], 0, 1);
                __builtin_prefetch(&v_cache[srcn], 0, 1);
            }
        }

        if (kvBase <= qRowMax) {
            // ---- S = Q * K^T  (two 16x16 tiles over 32 kv cols) ----
            v8f acc0 = {}, acc1 = {};
#pragma unroll
            for (int c = 0; c < 4; ++c) {
                const v16u bK0 = *(const v16u*)&sK[p][(0 * 16 + ln) * HDIM + c * 32 + lh * 16];
                const v16u bK1 = *(const v16u*)&sK[p][(1 * 16 + ln) * HDIM + c * 32 + lh * 16];
                acc0 = __builtin_amdgcn_wmma_f32_16x16x32_bf16(
                           false, as_bf(aQ[c]), false, as_bf(bK0), (short)0, acc0, false, false);
                acc1 = __builtin_amdgcn_wmma_f32_16x16x32_bf16(
                           false, as_bf(aQ[c]), false, as_bf(bK1), (short)0, acc1, false, false);
            }

            // ---- causal mask (diagonal tiles only; wave-uniform branch) ----
            if (kvBase + TILE_KV - 1 > qRowMin) {
#pragma unroll
                for (int e = 0; e < 8; ++e) {
                    const int qRow = qRowMin + lh * 8 + e;
                    if (kvBase + ln      > qRow) acc0[e] = -INFINITY;
                    if (kvBase + 16 + ln > qRow) acc1[e] = -INFINITY;
                }
            }

            // ---- online softmax (row max via ds_swizzle; sum kept per-lane) ----
            float p0[8], p1[8], corr[8];
#pragma unroll
            for (int e = 0; e < 8; ++e) {
                const float vmax = rowmax16(fmaxf(acc0[e], acc1[e]));
                const float mNew = fmaxf(mRow[e], vmax);
                corr[e] = __expf(mRow[e] - mNew);
                p0[e]   = __expf(acc0[e] - mNew);
                p1[e]   = __expf(acc1[e] - mNew);
                lRow[e] = lRow[e] * corr[e] + (p0[e] + p1[e]);   // per-lane partial
                mRow[e] = mNew;
            }
#pragma unroll
            for (int j = 0; j < 8; ++j)
#pragma unroll
                for (int e = 0; e < 8; ++e)
                    oAcc[j][e] *= corr[e];

            // ---- transpose P (C layout -> A layout) via wave-private LDS ----
#pragma unroll
            for (int e = 0; e < 8; ++e) {
                const int m = lh * 8 + e;
                sP[wave][m * TILE_KV + ln]      = f2bf(p0[e]);
                sP[wave][m * TILE_KV + 16 + ln] = f2bf(p1[e]);
            }
            const v8u lo = *(const v8u*)&sP[wave][ln * TILE_KV + lh * 16];
            const v8u hi = *(const v8u*)&sP[wave][ln * TILE_KV + lh * 16 + 8];
            v16u aP;
#pragma unroll
            for (int i = 0; i < 8; ++i) { aP[i] = lo[i]; aP[i + 8] = hi[i]; }

            // ---- O += P * V ----
#pragma unroll
            for (int j = 0; j < 8; ++j) {
                const v16u bV = *(const v16u*)&sVt[p][(j * 16 + ln) * VT_STRIDE + lh * 16];
                oAcc[j] = __builtin_amdgcn_wmma_f32_16x16x32_bf16(
                              false, as_bf(aP), false, as_bf(bV), (short)0, oAcc[j], false, false);
            }
        }

        // ---- drain next tile's loads into the other LDS buffer ----
        if (haveNext) {
#pragma unroll
            for (int i = 0; i < 4; ++i) {
                const int kv = rowG + 8 * i;
                *(v2ui*)&sK[1 - p][kv * HDIM + d0] = pack4bf(kN[i]);
#pragma unroll
                for (int dd = 0; dd < 4; ++dd)
                    sVt[1 - p][(d0 + dd) * VT_STRIDE + kv] = f2bf(vN[i][dd]);
            }
        }
        __syncthreads();
    }

    // ---- final cross-lane sum of l, normalize + store ----
#pragma unroll
    for (int e = 0; e < 8; ++e) {
        const float inv = 1.0f / rowsum16(lRow[e]);
        const int   tok = b * QLEN + qBase + wave * 16 + lh * 8 + e;
        float* orow = out + ((size_t)tok * NH + h) * HDIM + ln;
#pragma unroll
        for (int j = 0; j < 8; ++j)
            orow[j * 16] = oAcc[j][e] * inv;
    }
}

extern "C" void kernel_launch(void* const* d_in, const int* in_sizes, int n_in,
                              void* d_out, int out_size, void* d_ws, size_t ws_size,
                              hipStream_t stream) {
    const float* q            = (const float*)d_in[0];
    const float* k_cache      = (const float*)d_in[1];
    const float* v_cache      = (const float*)d_in[2];
    const int*   block_tables = (const int*)d_in[3];
    float*       out          = (float*)d_out;

    const int nBlocks = BATCH * NH * (QLEN / TILE_Q);   // 512
    qwen3_pa_fwd<<<dim3(nBlocks), dim3(256), 0, stream>>>(q, k_cache, v_cache,
                                                          block_tables, out);
}
